// TransformerBlock_20976620273903
// MI455X (gfx1250) — matrix operand order repaired and checked by reading, not compiled, not run on hardware
//
#include <hip/hip_runtime.h>
#include <cstdint>
#include <cstddef>

// ---------------------------------------------------------------------------
// Types for CDNA5 WMMA / TDM (gfx1250, wave32)
// ---------------------------------------------------------------------------
typedef __attribute__((ext_vector_type(16))) __bf16 v16bf;
typedef __attribute__((ext_vector_type(8)))  __bf16 bhalf8;   // 16 bytes
typedef __attribute__((ext_vector_type(4)))  __bf16 bhalf4;   // 8 bytes
typedef __attribute__((ext_vector_type(8)))  float  v8f;
typedef __attribute__((ext_vector_type(4)))  float  v4f;
typedef __attribute__((ext_vector_type(4)))  unsigned int u32x4;
typedef __attribute__((ext_vector_type(8)))  int i32x8;
typedef __attribute__((ext_vector_type(4)))  int i32x4;

union AF { v16bf v; bhalf8 h[2]; __bf16 e[16]; };

#define BATCH  4
#define SEQ    4096
#define DMODEL 1024
#define NHEAD  16
#define DHEAD  64
#define FFDIM  4096
#define MROWS  (BATCH*SEQ)      // 16384 token rows

__device__ __forceinline__ __bf16 f2bf(float f) { return (__bf16)f; }

// Low 32 bits of a generic pointer to __shared__ memory = LDS byte address.
__device__ __forceinline__ uint32_t lds_off(const void* p) {
  return (uint32_t)(uintptr_t)p;
}

// ---------------------------------------------------------------------------
// Tensor Data Mover: 2D tile load Global -> LDS (bf16, 2-byte elements).
// D# per CDNA5 ISA ch.8: group0 = {count/lds/global/type}, group1 = dims.
// pad_interval code 3 => pad after every 16 DWORDs (= 32 bf16 = one tile row)
// pad_amount  code 3 => insert 4 DWORDs (= 8 bf16)  -> LDS row pitch 40 bf16.
// NOTE: this toolchain exposes the 6-arg builtin (extra int32x8 group).
// ---------------------------------------------------------------------------
__device__ __forceinline__ void tdm_load_2d(uint32_t lds_addr, const void* gaddr,
                                            uint32_t tensor_d0, uint32_t tensor_d1,
                                            uint32_t tile_d0,   uint32_t tile_d1,
                                            uint32_t stride0_elems) {
  uint64_t ga = (uint64_t)(uintptr_t)gaddr;
  u32x4 g0;
  g0[0] = 1u;                                          // count=1, user mode
  g0[1] = lds_addr;                                    // lds_addr[31:0]
  g0[2] = (uint32_t)ga;                                // global_addr[31:0]
  g0[3] = (uint32_t)((ga >> 32) & 0x1FFFFFFu) | (2u << 30);  // [56:32] | type=2
  i32x8 g1;
  g1[0] = (int)((1u << 16) |        // data_size = 1 (2 bytes)
                (1u << 20) |        // pad_enable
                (3u << 22) |        // pad_interval: 16 DWORDs
                (3u << 25));        // pad_amount : 4 DWORDs
  g1[1] = (int)((tensor_d0 & 0xFFFFu) << 16);                       // dim0[15:0]
  g1[2] = (int)((tensor_d0 >> 16) | ((tensor_d1 & 0xFFFFu) << 16)); // dim0 hi | dim1 lo
  g1[3] = (int)((tensor_d1 >> 16) | (tile_d0 << 16));               // dim1 hi | tile0
  g1[4] = (int)(tile_d1 & 0xFFFFu);                                 // tile1 | tile2=0
  g1[5] = (int)stride0_elems;                                       // dim0_stride lo
  g1[6] = 0;
  g1[7] = 0;
  i32x4 z4 = {0, 0, 0, 0};
  i32x8 z8 = {0, 0, 0, 0, 0, 0, 0, 0};
  __builtin_amdgcn_tensor_load_to_lds(g0, g1, z4, z4, z8, 0);
}

// ---------------------------------------------------------------------------
// f32 -> bf16 (row-major, x4)
// ---------------------------------------------------------------------------
__global__ __launch_bounds__(256) void f32_to_bf16_k(const float* __restrict__ in,
                                                     __bf16* __restrict__ out, int n4) {
  int i = blockIdx.x * 256 + threadIdx.x;
  if (i < n4) {
    v4f x = *(const v4f*)(in + (size_t)i * 4);
    bhalf4 y;
    y[0] = f2bf(x[0]); y[1] = f2bf(x[1]); y[2] = f2bf(x[2]); y[3] = f2bf(x[3]);
    *(bhalf4*)(out + (size_t)i * 4) = y;
  }
}

// ---------------------------------------------------------------------------
// f32 [K x N] -> bf16 [N x K] transposed (weights: done once, reused 16384x)
// ---------------------------------------------------------------------------
__global__ __launch_bounds__(256) void transpose_f32_bf16_k(const float* __restrict__ in,
                                                            __bf16* __restrict__ out,
                                                            int K, int N) {
  __shared__ float t[32][33];
  int n0 = blockIdx.x * 32, k0 = blockIdx.y * 32;
  int tx = threadIdx.x & 31, ty = threadIdx.x >> 5;    // ty 0..7
#pragma unroll
  for (int r = 0; r < 4; ++r)
    t[ty + 8 * r][tx] = in[(size_t)(k0 + ty + 8 * r) * N + n0 + tx];
  __syncthreads();
#pragma unroll
  for (int r = 0; r < 4; ++r)
    out[(size_t)(n0 + ty + 8 * r) * K + k0 + tx] = f2bf(t[tx][ty + 8 * r]);
}

// ---------------------------------------------------------------------------
// GEMM: C[M,N] = A[M,K] @ Bt[N,K]^T (+bias)(+SiLU).  Both operands bf16
// row-major 2D tiles -> staged by TDM with double buffering.
// Block 256 thr = 8 waves; block tile 128x128; wave tile 32x64; K-step 32.
// A frag (16-bit A 16x32): lanes 0-15: K0-7,K16-23; lanes 16-31: K8-15,K24-31.
// B frag (16-bit B 32x16): col = lane&15; lanes 0-15: K0-7,K16-23; lanes 16-31: K8-15,K24-31.
// ---------------------------------------------------------------------------
template<bool BIAS, bool SILU, bool OUT_BF16>
__global__ __launch_bounds__(256)
void gemm_bf16(const __bf16* __restrict__ A, const __bf16* __restrict__ Bt,
               const float* __restrict__ bias,
               float* __restrict__ Cf, __bf16* __restrict__ Cb,
               int M, int N, int K)
{
  constexpr int BM = 128, BN = 128, BK = 32, LD = 40;  // 80B pitch (16B aligned)
  __shared__ __bf16 As[2][BM * LD];
  __shared__ __bf16 Bs[2][BN * LD];
  const int tid  = threadIdx.x;
  const int lane = tid & 31, wave = tid >> 5;
  const int wm = wave & 3, wn = wave >> 2;             // 4(M) x 2(N) wave grid
  const int m0 = blockIdx.y * BM, n0 = blockIdx.x * BN;
  const int wm0 = wm * 32, wn0 = wn * 64;
  const int mrow  = lane & 15;
  const int khalf = lane >> 4;
  v8f acc[2][4] = {};
  const int nsteps = K / BK;
  const __bf16* Ab = A  + (size_t)m0 * K;
  const __bf16* Bb = Bt + (size_t)n0 * K;

  if (tid == 0) {
    tdm_load_2d(lds_off(&As[0][0]), Ab, K, M, BK, BM, K);
    tdm_load_2d(lds_off(&Bs[0][0]), Bb, K, N, BK, BN, K);
  }
  for (int s = 0; s < nsteps; ++s) {
    const int buf = s & 1;
    if (tid == 0) {
      if (s + 1 < nsteps) {
        tdm_load_2d(lds_off(&As[buf ^ 1][0]), Ab + (size_t)(s + 1) * BK, K, M, BK, BM, K);
        tdm_load_2d(lds_off(&Bs[buf ^ 1][0]), Bb + (size_t)(s + 1) * BK, K, N, BK, BN, K);
        __builtin_amdgcn_s_wait_tensorcnt((short)2);   // current step's pair done
      } else {
        __builtin_amdgcn_s_wait_tensorcnt((short)0);
      }
    }
    __syncthreads();
#pragma unroll
    for (int i = 0; i < 2; ++i) {
      AF a;
      const __bf16* ap = &As[buf][(wm0 + i * 16 + mrow) * LD];
      a.h[0] = *(const bhalf8*)(ap + khalf * 8);
      a.h[1] = *(const bhalf8*)(ap + 16 + khalf * 8);
#pragma unroll
      for (int j = 0; j < 4; ++j) {
        AF b;
        const __bf16* bp = &Bs[buf][(wn0 + j * 16 + mrow) * LD];
        b.h[0] = *(const bhalf8*)(bp + khalf * 8);
        b.h[1] = *(const bhalf8*)(bp + 16 + khalf * 8);
        acc[i][j] = __builtin_amdgcn_wmma_f32_16x16x32_bf16(
            false, a.v, false, b.v, (short)0, acc[i][j], false, false);
      }
    }
    __syncthreads();   // protect buffer before it is re-filled
  }
  // epilogue: C layout VGPR r -> M = r + 8*khalf, N = lane&15
#pragma unroll
  for (int i = 0; i < 2; ++i) {
#pragma unroll
    for (int j = 0; j < 4; ++j) {
      int col = n0 + wn0 + j * 16 + mrow;
      float bv = BIAS ? bias[col] : 0.f;
#pragma unroll
      for (int r = 0; r < 8; ++r) {
        int row = m0 + wm0 + i * 16 + khalf * 8 + r;
        float v = acc[i][j][r] + bv;
        if (SILU) v = v / (1.f + __expf(-v));
        if (OUT_BF16) Cb[(size_t)row * N + col] = f2bf(v);
        else          Cf[(size_t)row * N + col] = v;
      }
    }
  }
}

// ---------------------------------------------------------------------------
// q softmax over head dim (64), scaled; output TRANSPOSED qs_t[bh][d][n] bf16
// ---------------------------------------------------------------------------
__global__ __launch_bounds__(256) void q_softmax_k(const float* __restrict__ qkv,
                                                   __bf16* __restrict__ qst) {
  const float scale = 0.125f;     // DH^-0.5
  int lane = threadIdx.x & 31, wave = threadIdx.x >> 5;
  int gw = blockIdx.x * 8 + wave;
  int stride = gridDim.x * 8;
  for (int idx = gw; idx < NHEAD * BATCH * SEQ; idx += stride) {
    int bh = idx >> 12, n = idx & (SEQ - 1);
    int b = bh >> 4, h = bh & 15;
    const float* p = qkv + ((size_t)b * SEQ + n) * (3 * DMODEL) + h * DHEAD + lane * 2;
    float2 v = *(const float2*)p;
    float a0 = v.x * scale, a1 = v.y * scale;
    float m = fmaxf(a0, a1);
#pragma unroll
    for (int s = 16; s > 0; s >>= 1) m = fmaxf(m, __shfl_xor(m, s, 32));
    float e0 = __expf(a0 - m), e1 = __expf(a1 - m);
    float ss = e0 + e1;
#pragma unroll
    for (int s = 16; s > 0; s >>= 1) ss += __shfl_xor(ss, s, 32);
    float r = 1.f / ss;
    size_t base = ((size_t)bh * DHEAD + lane * 2) * SEQ + n;
    qst[base]       = f2bf(e0 * r);
    qst[base + SEQ] = f2bf(e1 * r);
  }
}

// ---------------------------------------------------------------------------
// k softmax over sequence (n) per (b,h,d); output TRANSPOSED ks_t[bh][d][n]
// ---------------------------------------------------------------------------
__global__ __launch_bounds__(256) void k_softmax_k(const float* __restrict__ qkv,
                                                   __bf16* __restrict__ kst) {
  int bh = blockIdx.x;
  int b = bh >> 4, h = bh & 15;
  int d = threadIdx.x & 63, g = threadIdx.x >> 6;
  __shared__ float red[4][64];
  const float* base = qkv + (size_t)b * SEQ * (3 * DMODEL) + DMODEL + h * DHEAD + d;
  float m = -3.4e38f;
  for (int n = g; n < SEQ; n += 4) m = fmaxf(m, base[(size_t)n * (3 * DMODEL)]);
  red[g][d] = m; __syncthreads();
  m = fmaxf(fmaxf(red[0][d], red[1][d]), fmaxf(red[2][d], red[3][d]));
  __syncthreads();
  float s = 0.f;
  for (int n = g; n < SEQ; n += 4) s += __expf(base[(size_t)n * (3 * DMODEL)] - m);
  red[g][d] = s; __syncthreads();
  s = red[0][d] + red[1][d] + red[2][d] + red[3][d];
  float r = 1.f / s;
  __bf16* out = kst + ((size_t)bh * DHEAD + d) * SEQ;
  for (int n = g; n < SEQ; n += 4)
    out[n] = f2bf(__expf(base[(size_t)n * (3 * DMODEL)] - m) * r);
}

// ---------------------------------------------------------------------------
// ctx = q^T @ k per (b,h): 64x64, K=4096; q/k already transposed [bh][d][n]
// so both tiles are plain 2D strided tiles -> staged with TDM.
// Output stored TRANSPOSED ctxT[bh][e][d] for the v@ctx consumer.
// ---------------------------------------------------------------------------
__global__ __launch_bounds__(128) void ctx_gemm_k(const __bf16* __restrict__ qst,
                                                  const __bf16* __restrict__ kst,
                                                  __bf16* __restrict__ ctxT) {
  constexpr int LD = 40;
  __shared__ __bf16 Qt[64 * LD];   // Qt[d][n-chunk]
  __shared__ __bf16 Kt[64 * LD];   // Kt[e][n-chunk]
  int bh = blockIdx.x;
  int tid = threadIdx.x, lane = tid & 31, wave = tid >> 5;
  const __bf16* qb = qst + (size_t)bh * DHEAD * SEQ;
  const __bf16* kb = kst + (size_t)bh * DHEAD * SEQ;
  int mrow = lane & 15, khalf = lane >> 4;
  v8f acc[4] = {};
  for (int n0 = 0; n0 < SEQ; n0 += 32) {
    if (tid == 0) {
      tdm_load_2d(lds_off(Qt), qb + n0, SEQ, DHEAD, 32, 64, SEQ);
      tdm_load_2d(lds_off(Kt), kb + n0, SEQ, DHEAD, 32, 64, SEQ);
      __builtin_amdgcn_s_wait_tensorcnt((short)0);
    }
    __syncthreads();
    AF a;
    const __bf16* ap = &Qt[(wave * 16 + mrow) * LD];
    a.h[0] = *(const bhalf8*)(ap + khalf * 8);
    a.h[1] = *(const bhalf8*)(ap + 16 + khalf * 8);
#pragma unroll
    for (int j = 0; j < 4; ++j) {
      AF bb;
      const __bf16* bp = &Kt[(j * 16 + mrow) * LD];
      bb.h[0] = *(const bhalf8*)(bp + khalf * 8);
      bb.h[1] = *(const bhalf8*)(bp + 16 + khalf * 8);
      acc[j] = __builtin_amdgcn_wmma_f32_16x16x32_bf16(
          false, a.v, false, bb.v, (short)0, acc[j], false, false);
    }
    __syncthreads();
  }
#pragma unroll
  for (int j = 0; j < 4; ++j) {
    int e = j * 16 + mrow;
#pragma unroll
    for (int r = 0; r < 8; ++r) {
      int d = wave * 16 + khalf * 8 + r;
      ctxT[(size_t)bh * (DHEAD * DHEAD) + e * DHEAD + d] = f2bf(acc[j][r]);  // transposed
    }
  }
}

// ---------------------------------------------------------------------------
// out = v @ ctx per (b,h). v converted bf16 on the fly from qkv f32;
// B fragments read straight from global ctxT (8KB/head, L2 resident).
// Block 128 thr = 4 waves x 32 rows; grid (BH, SEQ/128).
// ---------------------------------------------------------------------------
__global__ __launch_bounds__(128) void attn_vctx_k(const float* __restrict__ qkv,
                                                   const __bf16* __restrict__ ctxT,
                                                   __bf16* __restrict__ outb) {
  int bh = blockIdx.x; int b = bh >> 4, h = bh & 15;
  int tid = threadIdx.x, lane = tid & 31, wave = tid >> 5;
  int n0 = blockIdx.y * 128 + wave * 32;
  int mrow = lane & 15, khalf = lane >> 4;
  v8f acc[2][4] = {};
  const float*  vbase = qkv + ((size_t)b * SEQ + n0) * (3 * DMODEL) + 2 * DMODEL + h * DHEAD;
  const __bf16* cb = ctxT + (size_t)bh * (DHEAD * DHEAD);
#pragma unroll
  for (int k0 = 0; k0 < DHEAD; k0 += 32) {
#pragma unroll
    for (int i = 0; i < 2; ++i) {
      const float* ap = vbase + (size_t)(i * 16 + mrow) * (3 * DMODEL);
      AF a;
#pragma unroll
      for (int sg = 0; sg < 2; ++sg) {
        int kb = k0 + sg * 16 + khalf * 8;
        v4f f0 = *(const v4f*)(ap + kb);
        v4f f1 = *(const v4f*)(ap + kb + 4);
#pragma unroll
        for (int j = 0; j < 4; ++j) { a.e[sg * 8 + j] = f2bf(f0[j]); a.e[sg * 8 + 4 + j] = f2bf(f1[j]); }
      }
#pragma unroll
      for (int j = 0; j < 4; ++j) {
        AF bb;
        const __bf16* bp = cb + (j * 16 + mrow) * DHEAD + k0;
        bb.h[0] = *(const bhalf8*)(bp + khalf * 8);
        bb.h[1] = *(const bhalf8*)(bp + 16 + khalf * 8);
        acc[i][j] = __builtin_amdgcn_wmma_f32_16x16x32_bf16(
            false, a.v, false, bb.v, (short)0, acc[i][j], false, false);
      }
    }
  }
#pragma unroll
  for (int i = 0; i < 2; ++i) {
#pragma unroll
    for (int j = 0; j < 4; ++j) {
      int col = h * DHEAD + j * 16 + mrow;
#pragma unroll
      for (int r = 0; r < 8; ++r) {
        size_t row = (size_t)b * SEQ + n0 + i * 16 + khalf * 8 + r;
        outb[row * DMODEL + col] = f2bf(acc[i][j][r]);
      }
    }
  }
}

// ---------------------------------------------------------------------------
// LayerNorm over D=1024 per row -> bf16
// ---------------------------------------------------------------------------
__global__ __launch_bounds__(256) void layernorm_k(const float* __restrict__ attn,
                                                   const float* __restrict__ g,
                                                   const float* __restrict__ be,
                                                   __bf16* __restrict__ hb) {
  int row = blockIdx.x, t = threadIdx.x;
  const float* p = attn + (size_t)row * DMODEL;
  v4f x = *(const v4f*)(p + t * 4);
  __shared__ float rbuf[8];
  float s = x[0] + x[1] + x[2] + x[3];
#pragma unroll
  for (int o = 16; o > 0; o >>= 1) s += __shfl_xor(s, o, 32);
  if ((t & 31) == 0) rbuf[t >> 5] = s;
  __syncthreads();
  float mean = (rbuf[0] + rbuf[1] + rbuf[2] + rbuf[3] +
                rbuf[4] + rbuf[5] + rbuf[6] + rbuf[7]) * (1.f / DMODEL);
  __syncthreads();
  float d0 = x[0] - mean, d1 = x[1] - mean, d2 = x[2] - mean, d3 = x[3] - mean;
  float vs = d0 * d0 + d1 * d1 + d2 * d2 + d3 * d3;
#pragma unroll
  for (int o = 16; o > 0; o >>= 1) vs += __shfl_xor(vs, o, 32);
  if ((t & 31) == 0) rbuf[t >> 5] = vs;
  __syncthreads();
  float var = (rbuf[0] + rbuf[1] + rbuf[2] + rbuf[3] +
               rbuf[4] + rbuf[5] + rbuf[6] + rbuf[7]) * (1.f / DMODEL);
  float rs = __frsqrt_rn(var + 1e-5f);
  int c = t * 4;
  bhalf4 y;
  y[0] = f2bf(d0 * rs * g[c + 0] + be[c + 0]);
  y[1] = f2bf(d1 * rs * g[c + 1] + be[c + 1]);
  y[2] = f2bf(d2 * rs * g[c + 2] + be[c + 2]);
  y[3] = f2bf(d3 * rs * g[c + 3] + be[c + 3]);
  *(bhalf4*)(hb + (size_t)row * DMODEL + c) = y;
}

// ---------------------------------------------------------------------------
// host launcher
// ---------------------------------------------------------------------------
extern "C" void kernel_launch(void* const* d_in, const int* in_sizes, int n_in,
                              void* d_out, int out_size, void* d_ws, size_t ws_size,
                              hipStream_t stream) {
  const float* x    = (const float*)d_in[0];
  const float* Wqkv = (const float*)d_in[1];
  const float* Wout = (const float*)d_in[2];
  const float* bout = (const float*)d_in[3];
  const float* lng  = (const float*)d_in[4];
  const float* lnb  = (const float*)d_in[5];
  const float* W1   = (const float*)d_in[6];
  const float* b1   = (const float*)d_in[7];
  const float* W2   = (const float*)d_in[8];
  const float* b2   = (const float*)d_in[9];
  (void)in_sizes; (void)n_in; (void)out_size; (void)ws_size;

  char* ws = (char*)d_ws;
  size_t off = 0;
  auto alloc = [&](size_t bytes) { size_t r = off; off += (bytes + 255) & ~(size_t)255; return r; };

  const size_t N_WQKV = (size_t)DMODEL * 3 * DMODEL;
  const size_t N_WOUT = (size_t)DMODEL * DMODEL;
  const size_t N_W1   = (size_t)DMODEL * FFDIM;
  const size_t N_W2   = (size_t)FFDIM * DMODEL;
  const size_t N_TOK  = (size_t)MROWS * DMODEL;
  const size_t N_QKV  = (size_t)MROWS * 3 * DMODEL;
  const size_t N_CTX  = (size_t)BATCH * NHEAD * DHEAD * DHEAD;

  size_t o_wqkv = alloc(N_WQKV * 2);       // transposed bf16 weights
  size_t o_wout = alloc(N_WOUT * 2);
  size_t o_w1   = alloc(N_W1 * 2);
  size_t o_w2   = alloc(N_W2 * 2);
  size_t o_xout = alloc(N_TOK * 2);        // x_bf16, later reused as out_bf16
  size_t o_qkv  = alloc(N_QKV * 4);        // qkv f32; later h_bf16 + y1_bf16
  size_t o_qsks = alloc(2 * N_TOK * 2);    // qs_t+ks_t bf16; later attn f32
  size_t o_ctx  = alloc(N_CTX * 2);

  __bf16* wqkv_t = (__bf16*)(ws + o_wqkv);
  __bf16* wout_t = (__bf16*)(ws + o_wout);
  __bf16* w1_t   = (__bf16*)(ws + o_w1);
  __bf16* w2_t   = (__bf16*)(ws + o_w2);
  __bf16* x_b    = (__bf16*)(ws + o_xout);
  __bf16* out_b  = (__bf16*)(ws + o_xout);                     // x dead after gemm1
  float*  qkv_f  = (float*)(ws + o_qkv);
  __bf16* h_b    = (__bf16*)(ws + o_qkv);                      // qkv dead by then
  __bf16* y1_b   = (__bf16*)(ws + o_qkv + N_TOK * 2);
  __bf16* qst_b  = (__bf16*)(ws + o_qsks);
  __bf16* kst_b  = (__bf16*)(ws + o_qsks + N_TOK * 2);
  float*  attn_f = (float*)(ws + o_qsks);                      // qs/ks dead by then
  __bf16* ctx_b  = (__bf16*)(ws + o_ctx);
  float*  y_out  = (float*)d_out;

  // activations: plain convert; weights: transpose-convert to [N x K]
  {
    int n4 = (int)(N_TOK / 4);
    f32_to_bf16_k<<<(n4 + 255) / 256, 256, 0, stream>>>(x, x_b, n4);
  }
  transpose_f32_bf16_k<<<dim3(3 * DMODEL / 32, DMODEL / 32), 256, 0, stream>>>(Wqkv, wqkv_t, DMODEL, 3 * DMODEL);
  transpose_f32_bf16_k<<<dim3(DMODEL / 32, DMODEL / 32),     256, 0, stream>>>(Wout, wout_t, DMODEL, DMODEL);
  transpose_f32_bf16_k<<<dim3(FFDIM / 32, DMODEL / 32),      256, 0, stream>>>(W1,   w1_t,   DMODEL, FFDIM);
  transpose_f32_bf16_k<<<dim3(DMODEL / 32, FFDIM / 32),      256, 0, stream>>>(W2,   w2_t,   FFDIM, DMODEL);

  // qkv = x @ W_qkv   [16384 x 3072]
  gemm_bf16<false,false,false><<<dim3(3 * DMODEL / 128, MROWS / 128), 256, 0, stream>>>(
      x_b, wqkv_t, nullptr, qkv_f, nullptr, MROWS, 3 * DMODEL, DMODEL);

  // dual softmax (outputs transposed [bh][d][n])
  q_softmax_k<<<2048, 256, 0, stream>>>(qkv_f, qst_b);
  k_softmax_k<<<BATCH * NHEAD, 256, 0, stream>>>(qkv_f, kst_b);

  // ctx = q^T k per head (64 heads, 64x64, K=4096) -> ctxT
  ctx_gemm_k<<<BATCH * NHEAD, 128, 0, stream>>>(qst_b, kst_b, ctx_b);

  // out = v @ ctx -> out_b [16384 x 1024] bf16
  attn_vctx_k<<<dim3(BATCH * NHEAD, SEQ / 128), 128, 0, stream>>>(qkv_f, ctx_b, out_b);

  // attn = out @ W_out + b_out  (f32)
  gemm_bf16<true,false,false><<<dim3(DMODEL / 128, MROWS / 128), 256, 0, stream>>>(
      out_b, wout_t, bout, attn_f, nullptr, MROWS, DMODEL, DMODEL);

  // LayerNorm -> h_b (bf16)
  layernorm_k<<<MROWS, 256, 0, stream>>>(attn_f, lng, lnb, h_b);

  // y1 = silu(h @ W1 + b1)  [16384 x 4096] bf16
  gemm_bf16<true,true,true><<<dim3(FFDIM / 128, MROWS / 128), 256, 0, stream>>>(
      h_b, w1_t, b1, nullptr, y1_b, MROWS, FFDIM, DMODEL);

  // y = y1 @ W2 + b2 -> d_out (f32)
  gemm_bf16<true,false,false><<<dim3(DMODEL / 128, MROWS / 128), 256, 0, stream>>>(
      y1_b, w2_t, b2, y_out, nullptr, MROWS, DMODEL, FFDIM);
}
